// MoELayer_84284438217169
// MI455X (gfx1250) — matrix operand order repaired and checked by reading, not compiled, not run on hardware
//
#include <hip/hip_runtime.h>
#include <hip/hip_bf16.h>

// ---------------------------------------------------------------------------
// MoE (top-2) for MI455X / gfx1250, wave32 + WMMA bf16 + async global->LDS.
//
//  - Router: 1 wave per token, shfl_xor reduction, softmax/top2, atomic append
//    to per-expert token lists (combine==0 off top-2, so top-2-only is exact).
//  - Experts, per (expert, 32-token tile) block:
//      X rows gathered with GLOBAL_LOAD_ASYNC_TO_LDS_B128 (ASYNCcnt) into
//      Xs (32x1024 bf16, 64KB LDS) --GEMM1+SiLU--> Hs (32x2048 bf16, 128KB LDS)
//      --GEMM2--> atomicAdd(out, weight * y).
//    192KB LDS / workgroup (CDNA5: 320KB per WGP).
//  - Each wave computes a 2x4 grid of 16x16 tiles per pass: A-frags reused 4x,
//    B-frags 2x -> 1.5 fragment loads per WMMA (vs 3 with 2x1 tiling).
//  - Weights converted fp32->bf16 once per launch into d_ws (67MB, fits the
//    192MB L2, so per-tile B re-reads stay on-chip).
//  - Workspace requirement: ~71.5 MB.
// ---------------------------------------------------------------------------

#define B_   2
#define S_   1024
#define D_   1024
#define E_   8
#define K_   2
#define FF_  2048
#define NTOK (B_ * S_)   // 2048 tokens
#define MT   32          // tokens per expert tile

typedef __attribute__((ext_vector_type(16))) __bf16 v16bf;
typedef __attribute__((ext_vector_type(8)))  __bf16 v8bf;
typedef __attribute__((ext_vector_type(8)))  float  v8f;

__device__ __forceinline__ unsigned short f32_to_bf16(float f) {
  unsigned u = __float_as_uint(f);
  u += 0x7fffu + ((u >> 16) & 1u);           // round-to-nearest-even
  return (unsigned short)(u >> 16);
}

__device__ __forceinline__ v16bf frag16(const __bf16* p0, const __bf16* p1) {
  v8bf lo = *(const v8bf*)p0;
  v8bf hi = *(const v8bf*)p1;
  return __builtin_shufflevector(lo, hi, 0,1,2,3,4,5,6,7,8,9,10,11,12,13,14,15);
}

// A-matrix 16x32 bf16 (ISA 7.12.2): lane<16 -> M=lane, K=0..7 & 16..23;
// lane>=16 -> M=lane-16, K=8..15 & 24..31.
__device__ __forceinline__ v16bf load_a(const __bf16* base, int ld, int lane, int kb) {
  int m = lane & 15, kh = lane >> 4;
  const __bf16* r = base + m * ld + kb + kh * 8;
  return frag16(r, r + 16);
}

// B-matrix 32x16 (KxN), B[k][n] = W[n][k] with W row-major (stride ld):
// lane<16 -> N=lane, K=0..7 & 16..23; lane>=16 -> N=lane-16, K=8..15 & 24..31.
__device__ __forceinline__ v16bf load_b(const __bf16* wrows, int ld, int lane, int kb) {
  int n = lane & 15, kh = lane >> 4;
  const __bf16* r = wrows + n * ld + kb + kh * 8;
  return frag16(r, r + 16);
}

#define WMMA_BF16(A, Bm, C) \
  __builtin_amdgcn_wmma_f32_16x16x32_bf16(false, (A), false, (Bm), (short)0, (C), false, false)

// --------------------------- init -----------------------------------------
__global__ void moe_init(float* __restrict__ out, int* __restrict__ cnt) {
  int i = blockIdx.x * blockDim.x + threadIdx.x;
  int stride = gridDim.x * blockDim.x;
  for (int j = i; j < NTOK * D_; j += stride) out[j] = 0.0f;
  if (i < E_) cnt[i] = 0;
}

// --------------------------- fp32 -> bf16 ----------------------------------
__global__ void cvt_bf16(const float* __restrict__ src,
                         unsigned short* __restrict__ dst, int n4) {
  int i = blockIdx.x * blockDim.x + threadIdx.x;
  int stride = gridDim.x * blockDim.x;
  for (int j = i; j < n4; j += stride) {
    float4 v = ((const float4*)src)[j];
    ushort4 r;
    r.x = f32_to_bf16(v.x); r.y = f32_to_bf16(v.y);
    r.z = f32_to_bf16(v.z); r.w = f32_to_bf16(v.w);
    ((ushort4*)dst)[j] = r;
  }
}

// --------------------------- router ----------------------------------------
__global__ __launch_bounds__(256) void moe_router(
    const float* __restrict__ x, const float* __restrict__ rw,
    float* __restrict__ probs_out, float* __restrict__ idx_out,
    float* __restrict__ wt_out, int* __restrict__ cnt,
    int* __restrict__ etok, float* __restrict__ ewt) {
  int lane = threadIdx.x & 31;
  int wave = threadIdx.x >> 5;
  int t = blockIdx.x * 8 + wave;                 // one wave32 per token

  float acc[E_];
#pragma unroll
  for (int e = 0; e < E_; ++e) acc[e] = 0.0f;

  const float* xr = x + (size_t)t * D_;
  for (int it = 0; it < D_ / 32; ++it) {
    float xv = xr[it * 32 + lane];
#pragma unroll
    for (int e = 0; e < E_; ++e) acc[e] += xv * rw[e * D_ + it * 32 + lane];
  }
#pragma unroll
  for (int e = 0; e < E_; ++e)
    for (int off = 16; off; off >>= 1) acc[e] += __shfl_xor(acc[e], off, 32);

  if (lane == 0) {
    float mx = acc[0];
    for (int e = 1; e < E_; ++e) mx = fmaxf(mx, acc[e]);
    float p[E_], s = 0.0f;
    for (int e = 0; e < E_; ++e) { p[e] = __expf(acc[e] - mx); s += p[e]; }
    float inv = 1.0f / s;
    for (int e = 0; e < E_; ++e) { p[e] *= inv; probs_out[t * E_ + e] = p[e]; }
    // top-2, ties -> lower index (strict >)
    int i1 = 0;
    for (int e = 1; e < E_; ++e) if (p[e] > p[i1]) i1 = e;
    int i2 = (i1 == 0) ? 1 : 0;
    for (int e = 0; e < E_; ++e) if (e != i1 && p[e] > p[i2]) i2 = e;
    float renorm = 1.0f / (p[i1] + p[i2]);
    float wa = p[i1] * renorm, wb = p[i2] * renorm;
    idx_out[t * K_ + 0] = (float)i1; idx_out[t * K_ + 1] = (float)i2;
    wt_out[t * K_ + 0] = wa;         wt_out[t * K_ + 1] = wb;
    int pos = atomicAdd(&cnt[i1], 1);
    etok[i1 * NTOK + pos] = t; ewt[i1 * NTOK + pos] = wa;
    pos = atomicAdd(&cnt[i2], 1);
    etok[i2 * NTOK + pos] = t; ewt[i2 * NTOK + pos] = wb;
  }
}

// --------------------------- fused expert MLP -------------------------------
__global__ __launch_bounds__(256) void moe_expert(
    const unsigned short* __restrict__ xb,    // [NTOK, D]  bf16
    const unsigned short* __restrict__ w1b,   // [E, FF, D] bf16
    const unsigned short* __restrict__ w2b,   // [E, D, FF] bf16
    const int* __restrict__ cnt, const int* __restrict__ etok,
    const float* __restrict__ ewt, float* __restrict__ out) {
  extern __shared__ unsigned short smem[];    // Xs[32*1024] ++ Hs[32*2048] = 192KB
  __shared__ int   toksS[MT];
  __shared__ float wtsS[MT];
  unsigned short* Xs = smem;
  unsigned short* Hs = smem + MT * D_;

  const int e = blockIdx.y;
  const int tile = blockIdx.x;
  const int c = cnt[e];
  if (tile * MT >= c) return;                 // empty tile for this expert

  const int tid = threadIdx.x;
  const int lane = tid & 31;
  const int wave = tid >> 5;

  if (tid < MT) {
    int slot = tile * MT + tid;
    if (slot < c) { toksS[tid] = etok[e * NTOK + slot]; wtsS[tid] = ewt[e * NTOK + slot]; }
    else          { toksS[tid] = 0;                     wtsS[tid] = 0.0f; }
  }
  __syncthreads();

  { // gather 32 token rows into LDS via async copies (no VGPR staging):
    // 8 threads/row, 16 x GLOBAL_LOAD_ASYNC_TO_LDS_B128 (16B) per thread.
    int row = tid >> 3, part = tid & 7;
    const unsigned short* src = xb + (size_t)toksS[row] * D_ + part * 128;
    unsigned short* dstp = Xs + row * D_ + part * 128;
    unsigned ldsoff = (unsigned)(uintptr_t)(void*)dstp;   // addr[31:0] = LDS offset
#pragma unroll
    for (int i = 0; i < 16; ++i) {
      asm volatile("global_load_async_to_lds_b128 %0, %1, off"
                   :: "v"(ldsoff + 16u * i), "v"(src + 8 * i) : "memory");
    }
    asm volatile("s_wait_asynccnt 0x0" ::: "memory");
  }
  __syncthreads();

  const __bf16* Xsb = (const __bf16*)Xs;
  const __bf16* Hsb = (const __bf16*)Hs;
  const __bf16* W1 = (const __bf16*)w1b + (size_t)e * FF_ * D_;
  const __bf16* W2 = (const __bf16*)w2b + (size_t)e * D_ * FF_;

  // ---- Phase 1: Hs = silu(Xs @ W1^T)  (32 x 2048), 8192 WMMAs/block ----
  // Each wave: 2 m-tiles x 4 ff-tiles per pass, 4 passes (128 ff-tiles total).
  for (int i = 0; i < 4; ++i) {
    int q = wave + 8 * i;                     // quad index 0..31
    int ffbase = q * 64;                      // tiles at ffbase + 16*t
    v8f acc[2][4];
#pragma unroll
    for (int t = 0; t < 4; ++t) { acc[0][t] = (v8f){0,0,0,0,0,0,0,0}; acc[1][t] = (v8f){0,0,0,0,0,0,0,0}; }
    for (int kc = 0; kc < D_ / 32; ++kc) {
      int kb = kc * 32;
      v16bf a0 = load_a(Xsb,           D_, lane, kb);
      v16bf a1 = load_a(Xsb + 16 * D_, D_, lane, kb);
#pragma unroll
      for (int t = 0; t < 4; ++t) {
        v16bf b = load_b(W1 + (size_t)(ffbase + 16 * t) * D_, D_, lane, kb);
        acc[0][t] = WMMA_BF16(a0, b, acc[0][t]);
        acc[1][t] = WMMA_BF16(a1, b, acc[1][t]);
      }
    }
    int n = lane & 15, mh = lane >> 4;        // C/D: lane=N, VGPR r -> M=r+8*mh
#pragma unroll
    for (int t = 0; t < 4; ++t) {
#pragma unroll
      for (int r = 0; r < 8; ++r) {
        float v0 = acc[0][t][r]; v0 = v0 / (1.0f + __expf(-v0));   // SiLU
        float v1 = acc[1][t][r]; v1 = v1 / (1.0f + __expf(-v1));
        int m0 = mh * 8 + r;
        Hs[m0 * FF_ + ffbase + 16 * t + n]        = f32_to_bf16(v0);
        Hs[(m0 + 16) * FF_ + ffbase + 16 * t + n] = f32_to_bf16(v1);
      }
    }
  }
  __syncthreads();

  // ---- Phase 2: out += diag(w) * (Hs @ W2^T)  (32 x 1024), 8192 WMMAs ----
  // Each wave: 2 m-tiles x 4 n-tiles per pass, 2 passes (64 n-tiles total).
  for (int i = 0; i < 2; ++i) {
    int q = wave + 8 * i;                     // quad index 0..15
    int nbase = q * 64;
    v8f acc[2][4];
#pragma unroll
    for (int t = 0; t < 4; ++t) { acc[0][t] = (v8f){0,0,0,0,0,0,0,0}; acc[1][t] = (v8f){0,0,0,0,0,0,0,0}; }
    for (int kc = 0; kc < FF_ / 32; ++kc) {
      int kb = kc * 32;
      v16bf a0 = load_a(Hsb,            FF_, lane, kb);
      v16bf a1 = load_a(Hsb + 16 * FF_, FF_, lane, kb);
#pragma unroll
      for (int t = 0; t < 4; ++t) {
        v16bf b = load_b(W2 + (size_t)(nbase + 16 * t) * FF_, FF_, lane, kb);
        acc[0][t] = WMMA_BF16(a0, b, acc[0][t]);
        acc[1][t] = WMMA_BF16(a1, b, acc[1][t]);
      }
    }
    int n = lane & 15, mh = lane >> 4;
#pragma unroll
    for (int t = 0; t < 4; ++t) {
#pragma unroll
      for (int r = 0; r < 8; ++r) {
        int m0 = mh * 8 + r, m1 = m0 + 16;
        atomicAdd(&out[(size_t)toksS[m0] * D_ + nbase + 16 * t + n], acc[0][t][r] * wtsS[m0]);
        atomicAdd(&out[(size_t)toksS[m1] * D_ + nbase + 16 * t + n], acc[1][t][r] * wtsS[m1]);
      }
    }
  }
}

// --------------------------- host launch ------------------------------------
extern "C" void kernel_launch(void* const* d_in, const int* in_sizes, int n_in,
                              void* d_out, int out_size, void* d_ws, size_t ws_size,
                              hipStream_t stream) {
  const float* x  = (const float*)d_in[0];   // [B,S,D]
  const float* rw = (const float*)d_in[1];   // [E,D]
  const float* w1 = (const float*)d_in[2];   // [E,FF,D]
  const float* w2 = (const float*)d_in[3];   // [E,D,FF]
  float* out = (float*)d_out;

  // workspace layout (~71.5 MB required)
  unsigned short* w1b = (unsigned short*)d_ws;                 // E*FF*D bf16
  unsigned short* w2b = w1b + (size_t)E_ * FF_ * D_;           // E*D*FF bf16
  unsigned short* xb  = w2b + (size_t)E_ * D_ * FF_;           // NTOK*D bf16
  int*   cnt  = (int*)(xb + (size_t)NTOK * D_);
  int*   etok = cnt + 16;                                      // E*NTOK ids
  float* ewt  = (float*)(etok + E_ * NTOK);                    // E*NTOK weights

  // d_out tuple layout: output | router_probs | topk_indices | topk_weights
  float* probs = out + (size_t)NTOK * D_;
  float* idxo  = probs + (size_t)NTOK * E_;
  float* wto   = idxo + (size_t)NTOK * K_;

  moe_init  <<<1024, 256, 0, stream>>>(out, cnt);
  cvt_bf16  <<<2048, 256, 0, stream>>>(w1, w1b, E_ * FF_ * D_ / 4);
  cvt_bf16  <<<2048, 256, 0, stream>>>(w2, w2b, E_ * D_ * FF_ / 4);
  cvt_bf16  <<<512,  256, 0, stream>>>(x,  xb,  NTOK * D_ / 4);
  moe_router<<<NTOK / 8, 256, 0, stream>>>(x, rw, probs, idxo, wto, cnt, etok, ewt);

  size_t lds = (size_t)(MT * D_ + MT * FF_) * sizeof(unsigned short); // 192 KB
  moe_expert<<<dim3(NTOK / MT, E_), 256, lds, stream>>>(xb, w1b, w2b, cnt, etok, ewt, out);
}
